// GraphAttentionLayer_81209241633570
// MI455X (gfx1250) — matrix operand order repaired and checked by reading, not compiled, not run on hardware
//
#include <hip/hip_runtime.h>
#include <hip/hip_bf16.h>

typedef __attribute__((ext_vector_type(16))) __bf16 v16bf;
typedef __attribute__((ext_vector_type(8)))  float  v8f;

#define D 128                 // D_IN == D_OUT
#define TM 64                 // rows of x per GEMM block
#define WSTRIDE 136           // padded K-stride (elements) for LDS tiles (16B aligned, bank-spread)
#define WT_BYTES (D * WSTRIDE * 2)   // 34816 B : transposed W tile in LDS
#define SX_BYTES (TM * WSTRIDE * 2)  // 17408 B : x tile in LDS
#define NEG_SLOPE 0.2f
#define ALPHA_LRELU 0.2f

#if defined(__gfx1250__) && __has_builtin(__builtin_amdgcn_global_load_async_to_lds_b128)
#define ASYNC_LDS 1
#endif

#ifdef ASYNC_LDS
typedef int v4i __attribute__((ext_vector_type(4)));
typedef __attribute__((address_space(1))) v4i g_v4i;   // global
typedef __attribute__((address_space(3))) v4i l_v4i;   // LDS
// 16-byte async DMA: global -> LDS (tracked by ASYNCcnt)
__device__ __forceinline__ void async_copy16(const void* g, void* l) {
    __builtin_amdgcn_global_load_async_to_lds_b128(
        (g_v4i*)(unsigned long long)(size_t)g,     // inttoptr into AS1
        (l_v4i*)(unsigned)(size_t)l,               // low 32 bits = LDS offset, AS3
        0, 0);
}
#endif

// ---------------------------------------------------------------- init
__global__ __launch_bounds__(256) void gat_init(float* out, unsigned* m_u,
                                                float* denom, int n) {
    long long i = (long long)blockIdx.x * blockDim.x + threadIdx.x;
    if (i < (long long)n * D) out[i] = 0.0f;
    if (i < n) { m_u[i] = 0u; denom[i] = 0.0f; }
}

// ---------------------------------------------------------------- prep: f32 -> bf16 (x row-major; W transposed + padded)
__global__ __launch_bounds__(256) void gat_prep(const float* __restrict__ x,
                                                const float* __restrict__ W,
                                                __bf16* __restrict__ xb,
                                                __bf16* __restrict__ Wtb,
                                                long long nx) {
    long long i = (long long)blockIdx.x * blockDim.x + threadIdx.x;
    if (i < nx) xb[i] = (__bf16)x[i];
    if (i < D * D) {
        int k = (int)(i >> 7), c = (int)(i & (D - 1));
        Wtb[c * WSTRIDE + k] = (__bf16)W[i];   // transposed: [col][k]
    }
}

// ---------------------------------------------------------------- GEMM h = x @ W  (bf16 WMMA, f32 accumulate)
__global__ __launch_bounds__(128) void gat_gemm(const __bf16* __restrict__ xb,
                                                const __bf16* __restrict__ Wtb,
                                                float* __restrict__ h, int nrows) {
    __shared__ __align__(16) char smem[WT_BYTES + SX_BYTES];   // 52224 B
    __bf16* sWt  = (__bf16*)smem;                // [col][k], stride WSTRIDE
    __bf16* sX   = (__bf16*)(smem + WT_BYTES);   // [row][k], stride WSTRIDE
    float*  sOut = (float*)smem;                 // reused after compute (32 KB)

    const int tid  = threadIdx.x;
    const int wave = tid >> 5;
    const int lane = tid & 31;
    const int row0 = blockIdx.x * TM;

#ifdef ASYNC_LDS
    // W tile: contiguous byte copy (padding bytes harmless)
    for (int i = tid; i < WT_BYTES / 16; i += 128)
        async_copy16((const char*)Wtb + i * 16, smem + i * 16);
    // x tile: 16B chunks, global stride 256B -> LDS stride 272B
    for (int i = tid; i < TM * 16; i += 128) {
        int r = i >> 4, c = i & 15;
        int gr = row0 + r; if (gr >= nrows) gr = nrows - 1;   // clamp; extra rows never stored
        async_copy16(xb + (size_t)gr * D + c * 8,
                     smem + WT_BYTES + r * (WSTRIDE * 2) + c * 16);
    }
#if __has_builtin(__builtin_amdgcn_s_wait_asynccnt)
    __builtin_amdgcn_s_wait_asynccnt(0);
#else
    asm volatile("s_wait_asynccnt 0x0" ::: "memory");
#endif
    __syncthreads();
#else
    for (int i = tid; i < WT_BYTES / 16; i += 128)
        ((uint4*)smem)[i] = ((const uint4*)Wtb)[i];
    for (int i = tid; i < TM * 16; i += 128) {
        int r = i >> 4, c = i & 15;
        int gr = row0 + r; if (gr >= nrows) gr = nrows - 1;
        *(uint4*)(smem + WT_BYTES + r * (WSTRIDE * 2) + c * 16) =
            *(const uint4*)(xb + (size_t)gr * D + c * 8);
    }
    __syncthreads();
#endif

    const int lm   = lane & 15;
    const int isHi = lane >> 4;
    const int mloc = wave * 16;      // this wave's 16-row strip

    union Frag { uint4 u[2]; v16bf v; };
    v8f acc[8] = {};

    #pragma unroll
    for (int k0 = 0; k0 < D; k0 += 32) {
        // A: 16x32 bf16. lanes 0-15: K=0..7,16..23 ; lanes 16-31: K=8..15,24..31
        Frag A;
        const __bf16* ap = sX + (mloc + lm) * WSTRIDE + k0 + isHi * 8;
        A.u[0] = *(const uint4*)ap;          // K 0..7   (+hi offset)
        A.u[1] = *(const uint4*)(ap + 16);   // K 16..23 (+hi offset)
        #pragma unroll
        for (int n = 0; n < 8; ++n) {
            // B: 32x16 bf16 from transposed W. lanes 0-15: K=0..7,16..23 ; 16-31: K=8..15,24..31
            Frag B;
            const __bf16* bp = sWt + (n * 16 + lm) * WSTRIDE + k0 + isHi * 8;
            B.u[0] = *(const uint4*)bp;
            B.u[1] = *(const uint4*)(bp + 16);
            acc[n] = __builtin_amdgcn_wmma_f32_16x16x32_bf16(
                false, A.v, false, B.v, (short)0, acc[n], false, false);
        }
    }

    __syncthreads();    // done reading sWt/sX; reuse as sOut
    #pragma unroll
    for (int n = 0; n < 8; ++n)
        #pragma unroll
        for (int r = 0; r < 8; ++r)
            sOut[(mloc + (isHi ? 8 + r : r)) * D + n * 16 + lm] = acc[n][r];
    __syncthreads();

    // coalesced float4 stores (512B per wave-instruction)
    for (int i = tid; i < TM * (D / 4); i += 128) {
        int r = i >> 5;
        if (row0 + r < nrows)
            *(float4*)(h + (size_t)(row0 + r) * D + (i & 31) * 4) = ((const float4*)sOut)[i];
    }
}

// ---------------------------------------------------------------- per-node attention dots (wave per node)
__global__ __launch_bounds__(256) void gat_rowdots(const float* __restrict__ h,
                                                   const float* __restrict__ a_src,
                                                   const float* __restrict__ a_dst,
                                                   float* s_src, float* s_dst, int n) {
    const int node = blockIdx.x * 8 + (threadIdx.x >> 5);
    const int lane = threadIdx.x & 31;
    if (node >= n) return;
    const float4 hv = ((const float4*)(h + (size_t)node * D))[lane];
    const float4 as = ((const float4*)a_src)[lane];
    const float4 ad = ((const float4*)a_dst)[lane];
    float ss = hv.x * as.x + hv.y * as.y + hv.z * as.z + hv.w * as.w;
    float sd = hv.x * ad.x + hv.y * ad.y + hv.z * ad.z + hv.w * ad.w;
    #pragma unroll
    for (int o = 16; o > 0; o >>= 1) {
        ss += __shfl_xor(ss, o, 32);
        sd += __shfl_xor(sd, o, 32);
    }
    if (lane == 0) { s_src[node] = ss; s_dst[node] = sd; }
}

// order-preserving float<->uint transforms for atomic max
__device__ __forceinline__ unsigned f2ord(float f) {
    unsigned u = __float_as_uint(f);
    return (u & 0x80000000u) ? ~u : (u | 0x80000000u);
}
__device__ __forceinline__ float ord2f(unsigned v) {
    return __uint_as_float((v & 0x80000000u) ? (v & 0x7FFFFFFFu) : ~v);
}

// ---------------------------------------------------------------- edge pass 1: logits + segment max
__global__ __launch_bounds__(256) void gat_pass1(const long long* __restrict__ src,
                                                 const long long* __restrict__ dst,
                                                 const float* __restrict__ s_src,
                                                 const float* __restrict__ s_dst,
                                                 float* e_arr, unsigned* m_u,
                                                 int E, int N) {
    const int i = blockIdx.x * blockDim.x + threadIdx.x;
    if (i >= E + N) return;
    int s, d;
    if (i < E) { s = (int)src[i]; d = (int)dst[i]; } else { s = d = i - E; }
    float e = s_src[s] + s_dst[d];
    e = (e > 0.0f) ? e : NEG_SLOPE * e;
    e_arr[i] = e;
    atomicMax(&m_u[d], f2ord(e));
}

// ---------------------------------------------------------------- edge pass 2: exp + segment sum
__global__ __launch_bounds__(256) void gat_pass2(const long long* __restrict__ src,
                                                 const long long* __restrict__ dst,
                                                 const unsigned* __restrict__ m_u,
                                                 float* e_arr, float* denom,
                                                 int E, int N) {
    const int i = blockIdx.x * blockDim.x + threadIdx.x;
    if (i >= E + N) return;
    const int d = (i < E) ? (int)dst[i] : (i - E);
    const float ex = expf(e_arr[i] - ord2f(m_u[d]));
    e_arr[i] = ex;
    atomicAdd(&denom[d], ex);
}

// ---------------------------------------------------------------- edge pass 3: weighted scatter-add (wave per edge)
__global__ __launch_bounds__(256) void gat_pass3(const long long* __restrict__ src,
                                                 const long long* __restrict__ dst,
                                                 const float* __restrict__ e_arr,
                                                 const float* __restrict__ denom,
                                                 const float* __restrict__ h,
                                                 float* out, int E, int N) {
    const int edge = blockIdx.x * 8 + (threadIdx.x >> 5);
    const int lane = threadIdx.x & 31;
    if (edge >= E + N) return;
    int s, d;
    if (edge < E) { s = (int)src[edge]; d = (int)dst[edge]; } else { s = d = edge - E; }
    const float alpha = e_arr[edge] / denom[d];
    const float4 hv = ((const float4*)(h + (size_t)s * D))[lane];
    float* o = out + (size_t)d * D + lane * 4;
    atomicAdd(o + 0, alpha * hv.x);
    atomicAdd(o + 1, alpha * hv.y);
    atomicAdd(o + 2, alpha * hv.z);
    atomicAdd(o + 3, alpha * hv.w);
}

// ---------------------------------------------------------------- finalize: bias + LeakyReLU
__global__ __launch_bounds__(256) void gat_final(float* out, const float* __restrict__ bias,
                                                 int n) {
    long long i = (long long)blockIdx.x * blockDim.x + threadIdx.x;
    if (i >= (long long)n * D) return;
    float v = out[i] + bias[i & (D - 1)];
    out[i] = (v > 0.0f) ? v : ALPHA_LRELU * v;
}

// ---------------------------------------------------------------- launch
extern "C" void kernel_launch(void* const* d_in, const int* in_sizes, int n_in,
                              void* d_out, int out_size, void* d_ws, size_t ws_size,
                              hipStream_t stream) {
    const float*     x     = (const float*)d_in[0];
    const long long* ei    = (const long long*)d_in[1];   // int64 edge_index [2,E]
    const float*     W     = (const float*)d_in[2];
    const float*     a_src = (const float*)d_in[3];
    const float*     a_dst = (const float*)d_in[4];
    const float*     bias  = (const float*)d_in[5];
    float*           out   = (float*)d_out;

    const int N = in_sizes[0] / D;
    const int E = in_sizes[1] / 2;
    const long long* srcI = ei;
    const long long* dstI = ei + E;

    // workspace carve-up
    char* ws = (char*)d_ws;
    size_t off = 0;
    auto carve = [&](size_t bytes) { void* p = ws + off; off = (off + bytes + 255) & ~(size_t)255; return p; };
    float*    h     = (float*)   carve((size_t)N * D * 4);
    __bf16*   xb    = (__bf16*)  carve((size_t)N * D * 2);
    __bf16*   Wtb   = (__bf16*)  carve((size_t)WT_BYTES);
    float*    sS    = (float*)   carve((size_t)N * 4);
    float*    sD    = (float*)   carve((size_t)N * 4);
    unsigned* m_u   = (unsigned*)carve((size_t)N * 4);
    float*    denom = (float*)   carve((size_t)N * 4);
    float*    e_arr = (float*)   carve((size_t)(E + N) * 4);
    (void)ws_size;

    const long long tot  = (long long)N * D;
    const int       totE = E + N;

    gat_init<<<(int)((tot + 255) / 256), 256, 0, stream>>>(out, m_u, denom, N);
    gat_prep<<<(int)((tot + 255) / 256), 256, 0, stream>>>(x, W, xb, Wtb, tot);
    gat_gemm<<<(N + TM - 1) / TM, 128, 0, stream>>>(xb, Wtb, h, N);
    gat_rowdots<<<(N + 7) / 8, 256, 0, stream>>>(h, a_src, a_dst, sS, sD, N);
    gat_pass1<<<(totE + 255) / 256, 256, 0, stream>>>(srcI, dstI, sS, sD, e_arr, m_u, E, N);
    gat_pass2<<<(totE + 255) / 256, 256, 0, stream>>>(srcI, dstI, m_u, e_arr, denom, E, N);
    gat_pass3<<<(totE + 7) / 8, 256, 0, stream>>>(srcI, dstI, e_arr, denom, h, out, E, N);
    gat_final<<<(int)((tot + 255) / 256), 256, 0, stream>>>(out, bias, N);
}
